// L3MLcEmbedding_41034117546149
// MI455X (gfx1250) — matrix operand order repaired and checked by reading, not compiled, not run on hardware
//
#include <hip/hip_runtime.h>
#include <hip/hip_bf16.h>
#include <math.h>

#define HIDDEN 2048
#define SEQ    4096
#define VOCAB  32000
#define IDIM   64

typedef __attribute__((ext_vector_type(2)))  float    v2f;
typedef __attribute__((ext_vector_type(8)))  float    v8f;
typedef __attribute__((ext_vector_type(16))) _Float16 v16h;

__device__ __forceinline__ float gelu_exact(float x) {
    // exact (erf) GELU, matching torch nn.GELU() default
    return 0.5f * x * (1.0f + erff(x * 0.7071067811865475f));
}

// ---------------------------------------------------------------------------
// Kernel A: embedding gather  out[row] = wte[clamp(ids[row])]
// One float4 (16B) per thread -> global_load_b128 / global_store_b128 streams.
// Pure HBM bandwidth: ~268 MB moved => ~12 us at 23.3 TB/s.
// ---------------------------------------------------------------------------
__global__ __launch_bounds__(256) void embed_gather(const int* __restrict__ ids,
                                                    const float* __restrict__ wte,
                                                    float* __restrict__ out,
                                                    int n_rows) {
    const int vec_per_row = HIDDEN / 4;
    long idx = (long)blockIdx.x * blockDim.x + threadIdx.x;
    long total = (long)n_rows * vec_per_row;
    if (idx >= total) return;
    int row = (int)(idx / vec_per_row);
    int col = (int)(idx % vec_per_row);
    int id = ids[row];
    id = min(max(id, 0), VOCAB - 1);
    const float4* src = (const float4*)(wte + (long)id * HIDDEN);
    float4*       dst = (float4*)(out + (long)row * HIDDEN);
    dst[col] = src[col];
}

// ---------------------------------------------------------------------------
// Kernel B: lc-connector MLP + scatter-overwrite.
// 128 threads (4 waves), 16 LC rows per workgroup.
//   h0 = gelu(lc*W0 + b0)                      (scalar, LDS)
//   h1 = gelu(h0 @ W1 + b1)  -> v_wmma_f32_16x16x4_f32   (full fp32)
//   h2 = gelu(h1 @ W2 + b2)  -> v_wmma_f32_16x16x4_f32
//   o  = h2 @ Wout + bout -> v_wmma_f32_16x16x32_f16 (f32 accum)
// o rows stored directly at (pos_b*SEQ + pos_s), overwriting the gather.
// ---------------------------------------------------------------------------
__global__ __launch_bounds__(128) void lc_mlp_scatter(
    const float* __restrict__ lc,
    const int*   __restrict__ pos_b,
    const int*   __restrict__ pos_s,
    const float* __restrict__ W0,  const float* __restrict__ b0,
    const float* __restrict__ W1,  const float* __restrict__ b1,
    const float* __restrict__ W2,  const float* __restrict__ b2,
    const float* __restrict__ Wout, const float* __restrict__ bout,
    float* __restrict__ out)
{
    __shared__ float shA[16][IDIM + 4];   // +4 pad: conflict-free column reads
    __shared__ float shB[16][IDIM + 4];
    __shared__ long  shDest[16];

    const int tid  = threadIdx.x;
    const int wave = tid >> 5;            // wave32
    const int lane = tid & 31;
    const int lh   = lane >> 4;           // lane half (0: lanes 0-15, 1: 16-31)
    const int nl   = lane & 15;
    const int g0   = blockIdx.x * 16;     // first LC row of this workgroup

    if (tid < 16) {
        int g = g0 + tid;
        shDest[tid] = ((long)pos_b[g] * SEQ + pos_s[g]) * (long)HIDDEN;
    }

    // h0 = gelu(lc * W0 + b0): 16x64 elements, 8 per thread
    for (int e = tid; e < 16 * IDIM; e += 128) {
        int i = e >> 6, j = e & 63;
        shA[i][j] = gelu_exact(lc[g0 + i] * W0[j] + b0[j]);
    }
    __syncthreads();

    const int n0 = wave * 16;             // 16-column slab per wave (layers 1-2)

    // ----- layer 1: shB = gelu(shA @ W1 + b1) -----
    {
        v8f c = {};
        #pragma unroll
        for (int k = 0; k < IDIM; k += 4) {
            // f32 WMMA layouts (ISA 7.12.2):
            //  A 16x4:  VGPR0 = K=k+{0|2}, VGPR1 = K=k+{1|3}, M = lane%16
            //  B 4x16:  VGPR0 = K=k+{0|2}, VGPR1 = K=k+{1|3}, N = lane%16
            int kk = k + (lh ? 2 : 0);
            v2f a, b;
            a.x = shA[nl][kk];
            a.y = shA[nl][kk + 1];
            b.x = W1[kk * IDIM + n0 + nl];
            b.y = W1[(kk + 1) * IDIM + n0 + nl];
            c = __builtin_amdgcn_wmma_f32_16x16x4_f32(false, a, false, b,
                                                      (short)0, c, false, false);
        }
        float bias = b1[n0 + nl];
        #pragma unroll
        for (int r = 0; r < 8; ++r) {
            int M = r + (lh ? 8 : 0);     // C/D: VGPR r = M=r (lo half) / r+8 (hi)
            shB[M][n0 + nl] = gelu_exact(c[r] + bias);
        }
    }
    __syncthreads();

    // ----- layer 2: shA = gelu(shB @ W2 + b2) -----
    {
        v8f c = {};
        #pragma unroll
        for (int k = 0; k < IDIM; k += 4) {
            int kk = k + (lh ? 2 : 0);
            v2f a, b;
            a.x = shB[nl][kk];
            a.y = shB[nl][kk + 1];
            b.x = W2[kk * IDIM + n0 + nl];
            b.y = W2[(kk + 1) * IDIM + n0 + nl];
            c = __builtin_amdgcn_wmma_f32_16x16x4_f32(false, a, false, b,
                                                      (short)0, c, false, false);
        }
        float bias = b2[n0 + nl];
        #pragma unroll
        for (int r = 0; r < 8; ++r) {
            int M = r + (lh ? 8 : 0);
            shA[M][n0 + nl] = gelu_exact(c[r] + bias);
        }
    }
    __syncthreads();

    // ----- layer 3: out = shA @ Wout + bout, stored at scattered rows -----
    // Build f16 A fragments once (reused across all 32 N-tiles).
    // 16-bit A 16x32 layout: v<4 -> K = lh*8 + 2v+h ; v>=4 -> K = 16 + lh*8 + 2(v-4)+h
    v16h a0, a1;
    #pragma unroll
    for (int e = 0; e < 16; ++e) {
        int v = e >> 1, h = e & 1;
        int kl = (v < 4) ? (lh * 8 + 2 * v + h)
                         : (16 + lh * 8 + 2 * (v - 4) + h);
        a0[e] = (_Float16)shA[nl][kl];
        a1[e] = (_Float16)shA[nl][32 + kl];
    }

    const int colBase = wave * (HIDDEN / 4);       // 512 columns per wave
    for (int t = 0; t < HIDDEN / 4; t += 16) {
        const int nt = colBase + t;
        // 16-bit B 32x16 layout: v<4 -> K = lh*8 + 2v+h ; v>=4 -> K = 16 + lh*8 + 2(v-4)+h, N = lane%16
        v16h bf0, bf1;
        #pragma unroll
        for (int e = 0; e < 16; ++e) {
            int v = e >> 1, h = e & 1;
            int k = (v < 4) ? (lh * 8 + 2 * v + h) : (16 + lh * 8 + 2 * (v - 4) + h);
            bf0[e] = (_Float16)Wout[k * HIDDEN + nt + nl];
            bf1[e] = (_Float16)Wout[(32 + k) * HIDDEN + nt + nl];
        }
        v8f c = {};
        c = __builtin_amdgcn_wmma_f32_16x16x32_f16(false, a0, false, bf0,
                                                   (short)0, c, false, false);
        c = __builtin_amdgcn_wmma_f32_16x16x32_f16(false, a1, false, bf1,
                                                   (short)0, c, false, false);
        float bias = bout[nt + nl];
        #pragma unroll
        for (int r = 0; r < 8; ++r) {
            int M = r + (lh ? 8 : 0);
            out[shDest[M] + nt + nl] = c[r] + bias;
        }
    }
}

extern "C" void kernel_launch(void* const* d_in, const int* in_sizes, int n_in,
                              void* d_out, int out_size, void* d_ws, size_t ws_size,
                              hipStream_t stream) {
    const int*   ids   = (const int*)  d_in[0];
    const float* lc    = (const float*)d_in[1];
    const int*   pos_b = (const int*)  d_in[2];
    const int*   pos_s = (const int*)  d_in[3];
    const float* wte   = (const float*)d_in[4];
    const float* W0    = (const float*)d_in[5];
    const float* b0    = (const float*)d_in[6];
    const float* W1    = (const float*)d_in[7];
    const float* b1    = (const float*)d_in[8];
    const float* W2    = (const float*)d_in[9];
    const float* b2    = (const float*)d_in[10];
    const float* Wout  = (const float*)d_in[11];
    const float* bout  = (const float*)d_in[12];
    float* out = (float*)d_out;

    const int n_rows = in_sizes[0];                       // B*S
    long total_vec = (long)n_rows * (HIDDEN / 4);
    int blocks = (int)((total_vec + 255) / 256);
    embed_gather<<<blocks, 256, 0, stream>>>(ids, wte, out, n_rows);

    const int n_lc = in_sizes[1];                         // N_LC (lc_values is [N,1])
    lc_mlp_scatter<<<n_lc / 16, 128, 0, stream>>>(lc, pos_b, pos_s,
                                                  W0, b0, W1, b1, W2, b2,
                                                  Wout, bout, out);
}
